// DecoderMoEModel_56435870269506
// MI455X (gfx1250) — matrix operand order repaired and checked by reading, not compiled, not run on hardware
//
#include <hip/hip_runtime.h>
#include <hip/hip_bf16.h>
#include <stdint.h>
#include <stddef.h>

// ---------------------------------------------------------------------------
// DecoderMoE forward for MI455X (gfx1250).
// - All large GEMMs via v_wmma_f32_16x16x32_bf16 (wave32 WMMA, f32 accum),
//   64x128 block tile, 32x32 wave tile (4 WMMAs/wave/K-step).
// - Global->LDS staging via GLOBAL_LOAD_ASYNC_TO_LDS_B128 (ASYNCcnt) where
//   the source is contiguous; s_wait_asynccnt 0 before the consuming barrier.
// ---------------------------------------------------------------------------

#define USE_ASYNC 1

typedef __attribute__((ext_vector_type(16))) __bf16 v16bf;
typedef __attribute__((ext_vector_type(8)))  float  v8f;

#define VOCAB   32000
#define DMODEL  1024
#define NHEAD   16
#define NLAYER  2
#define HIDDEN  2048
#define NEXP    8
#define BATCH   2
#define SEQ     1024
#define HEADDIM 64
#define RHALF   32
#define NTOK    (BATCH * SEQ)

__device__ __forceinline__ unsigned short f2bf(float f) {
  unsigned u = __float_as_uint(f);
  unsigned r = 0x7FFFu + ((u >> 16) & 1u);   // round-to-nearest-even
  return (unsigned short)((u + r) >> 16);
}
__device__ __forceinline__ float bf2f(unsigned short h) {
  return __uint_as_float(((unsigned)h) << 16);
}
__device__ __forceinline__ float waveSum(float v) {
#pragma unroll
  for (int o = 16; o > 0; o >>= 1) v += __shfl_xor(v, o, 32);
  return v;
}

// LDS byte offset of a shared-memory pointer: for LDS, flat addr[31:0] is the
// LDS offset (CDNA5 aperture layout), so truncation is exact.
__device__ __forceinline__ unsigned ldsOff(const void* p) {
  return (unsigned)(unsigned long long)p;
}
// Async 16-byte global->LDS copy (GV mode, per-lane addresses).
__device__ __forceinline__ void asyncCopy16(unsigned lds_off,
                                            const unsigned short* gptr) {
#if USE_ASYNC
  asm volatile("global_load_async_to_lds_b128 %0, %1, off"
               :: "v"(lds_off), "v"((unsigned long long)gptr)
               : "memory");
#else
  *(uint4*)((char*)0 + 0) = *(const uint4*)gptr; // unused
#endif
}
__device__ __forceinline__ void asyncWait0() {
#if USE_ASYNC
  asm volatile("s_wait_asynccnt 0x0" ::: "memory");
#endif
}

// ---------------------------------------------------------------------------
// f32 -> bf16 cast
// ---------------------------------------------------------------------------
__global__ void cast_bf16_kernel(const float* __restrict__ src,
                                 unsigned short* __restrict__ dst, size_t n) {
  size_t i = (size_t)blockIdx.x * blockDim.x + threadIdx.x;
  size_t stride = (size_t)gridDim.x * blockDim.x;
  for (; i < n; i += stride) dst[i] = f2bf(src[i]);
}

// ---------------------------------------------------------------------------
// x = embed[tokens]   (f32)
// ---------------------------------------------------------------------------
__global__ void embed_gather_kernel(const int* __restrict__ tokens,
                                    const float* __restrict__ embed,
                                    float* __restrict__ x) {
  int t = blockIdx.x;
  int tok = tokens[t];
  const float* src = embed + (size_t)tok * DMODEL;
  float* dst = x + (size_t)t * DMODEL;
  for (int d = threadIdx.x; d < DMODEL; d += blockDim.x) dst[d] = src[d];
}

// ---------------------------------------------------------------------------
// RMSNorm -> bf16 (block per row)
// ---------------------------------------------------------------------------
__global__ void rmsnorm_bf16_kernel(const float* __restrict__ x,
                                    const float* __restrict__ w,
                                    unsigned short* __restrict__ out) {
  int row = blockIdx.x;
  const float* xr = x + (size_t)row * DMODEL;
  int tid = threadIdx.x, lane = tid & 31, wave = tid >> 5;
  float ss = 0.f;
  for (int d = tid; d < DMODEL; d += blockDim.x) { float v = xr[d]; ss += v * v; }
  ss = waveSum(ss);
  __shared__ float red[8];
  if (lane == 0) red[wave] = ss;
  __syncthreads();
  if (wave == 0) {
    float v = (lane < (int)(blockDim.x >> 5)) ? red[lane] : 0.f;
    v = waveSum(v);
    if (lane == 0) red[0] = v;
  }
  __syncthreads();
  float rs = rsqrtf(red[0] / (float)DMODEL + 1e-6f);
  unsigned short* o = out + (size_t)row * DMODEL;
  for (int d = tid; d < DMODEL; d += blockDim.x) o[d] = f2bf(xr[d] * rs * w[d]);
}

// ---------------------------------------------------------------------------
// WMMA GEMM: C[M,N] (+)= A[M,K] @ B[K,N]; A/B bf16, C f32.
// TRANSB: B stored [N,K] row-major (logits = x @ embed.T).
// MODE 0: C = AB ; 1: C += AB ; 2: C += AB * scale[row*sstride]
// 256 thr = 8 waves (2M x 4N); block tile 64x128; wave tile 32x32; K-step 32.
// Fragment layouts per CDNA5 ISA 7.12.2.
// ---------------------------------------------------------------------------
template <int MODE, bool TRANSB>
__global__ void gemm_wmma_kernel(const unsigned short* __restrict__ A,
                                 const unsigned short* __restrict__ B,
                                 float* __restrict__ C,
                                 int M, int N, int K,
                                 const float* __restrict__ scale, int sstride) {
  __shared__ __align__(16) unsigned As32[64 * 16];   // 64 m  x 32 k (bf16)
  __shared__ __align__(16) unsigned Bs32[128 * 16];  // 128 n x 32 k (n-major)
  const int n0 = blockIdx.x * 128;
  const int m0 = blockIdx.y * 64;
  const int t = threadIdx.x;
  const int lane = t & 31;
  const int wave = t >> 5;
  const int waveM = wave >> 2;   // 0..1
  const int waveN = wave & 3;    // 0..3
  const int mrow = lane & 15;
  const int half = lane >> 4;
  v8f acc00 = {}, acc01 = {}, acc10 = {}, acc11 = {};

  for (int k0 = 0; k0 < K; k0 += 32) {
    __syncthreads();
    {  // A tile 64x32: 16B per thread, contiguous -> async b128
      const int r = t >> 2;
      const int cp = t & 3;
#if USE_ASYNC
      asyncCopy16(ldsOff(&As32[r * 16 + cp * 4]),
                  A + (size_t)(m0 + r) * K + k0 + cp * 8);
#else
      const uint4* g = (const uint4*)(A + (size_t)(m0 + r) * K + k0);
      *(uint4*)&As32[r * 16 + cp * 4] = g[cp];
#endif
      if (k0 + 32 < K)
        __builtin_prefetch(A + (size_t)(m0 + r) * K + k0 + 32 + cp * 8, 0, 0);
    }
    if (TRANSB) {  // B[N,K]: rows contiguous in K -> async b128 x2 per thread
      const int n = t & 127;
      const int kh = (t >> 7) * 16;
      const unsigned short* src = B + (size_t)(n0 + n) * K + k0 + kh;
#if USE_ASYNC
      asyncCopy16(ldsOff(&Bs32[n * 16 + (kh >> 1)]), src);
      asyncCopy16(ldsOff(&Bs32[n * 16 + (kh >> 1) + 4]), src + 8);
#else
      const uint4* g = (const uint4*)src;
      *(uint4*)&Bs32[n * 16 + (kh >> 1)] = g[0];
      *(uint4*)&Bs32[n * 16 + (kh >> 1) + 4] = g[1];
#endif
    } else {  // B[K,N] row-major: transpose into n-major LDS (VALU scatter)
      unsigned short* Bs16 = (unsigned short*)Bs32;
      const int krow = t >> 4;       // 0..15 (handles krow and krow+16)
      const int n8 = (t & 15) * 8;
#pragma unroll
      for (int kk = 0; kk < 2; ++kk) {
        const int kr = krow + kk * 16;
        union { uint4 q; unsigned short s[8]; } u;
        u.q = *(const uint4*)(B + (size_t)(k0 + kr) * N + n0 + n8);
#pragma unroll
        for (int i = 0; i < 8; ++i) Bs16[(n8 + i) * 32 + kr] = u.s[i];
      }
      if (k0 + 32 < K)
        __builtin_prefetch(B + (size_t)(k0 + 32 + krow) * N + n0 + n8, 0, 0);
    }
    asyncWait0();
    __syncthreads();

    // Fragments (ISA 7.12.2 16-bit layouts) + 4 WMMAs per wave
    union FragU { v16bf v; unsigned u[8]; };
    FragU af[2], bfr[2];
#pragma unroll
    for (int mi = 0; mi < 2; ++mi) {
      const int rb = waveM * 32 + mi * 16 + mrow;
#pragma unroll
      for (int vv = 0; vv < 8; ++vv) {
        int ka = (vv < 4 ? vv * 2 : 16 + (vv - 4) * 2) + half * 8;
        af[mi].u[vv] = As32[rb * 16 + (ka >> 1)];
      }
    }
#pragma unroll
    for (int ni = 0; ni < 2; ++ni) {
      const int nb = waveN * 32 + ni * 16 + mrow;
#pragma unroll
      for (int vv = 0; vv < 8; ++vv)
        bfr[ni].u[vv] = Bs32[nb * 16 + half * 4 + (vv < 4 ? vv : vv + 4)];
    }
    acc00 = __builtin_amdgcn_wmma_f32_16x16x32_bf16(false, af[0].v, false, bfr[0].v, (short)0, acc00, false, false);
    acc01 = __builtin_amdgcn_wmma_f32_16x16x32_bf16(false, af[0].v, false, bfr[1].v, (short)0, acc01, false, false);
    acc10 = __builtin_amdgcn_wmma_f32_16x16x32_bf16(false, af[1].v, false, bfr[0].v, (short)0, acc10, false, false);
    acc11 = __builtin_amdgcn_wmma_f32_16x16x32_bf16(false, af[1].v, false, bfr[1].v, (short)0, acc11, false, false);
  }

  // C/D layout: vgpr r -> row = r + half*8 ; col = lane%16
  const int rowBase = m0 + waveM * 32;
  const int colBase = n0 + waveN * 32;
#pragma unroll
  for (int mi = 0; mi < 2; ++mi) {
#pragma unroll
    for (int ni = 0; ni < 2; ++ni) {
      v8f a = (mi == 0) ? (ni == 0 ? acc00 : acc01)
                        : (ni == 0 ? acc10 : acc11);
      const int col = colBase + ni * 16 + mrow;
#pragma unroll
      for (int r = 0; r < 8; ++r) {
        int row = rowBase + mi * 16 + r + half * 8;
        size_t idx = (size_t)row * N + col;
        float v = a[r];
        if (MODE == 0) C[idx] = v;
        else if (MODE == 1) C[idx] += v;
        else C[idx] += v * scale[(size_t)row * sstride];
      }
    }
  }
}

// ---------------------------------------------------------------------------
// RoPE in-place on q or k (f32): one thread per (token, head, i<32)
// ---------------------------------------------------------------------------
__global__ void rope_kernel(float* __restrict__ x) {
  int idx = blockIdx.x * blockDim.x + threadIdx.x;
  if (idx >= NTOK * NHEAD * RHALF) return;
  int i = idx & (RHALF - 1);
  int hh = (idx >> 5) & (NHEAD - 1);
  int tt = idx >> 9;                 // / (RHALF*NHEAD)
  int pos = tt & (SEQ - 1);
  float inv = __powf(10000.0f, -(float)i / (float)RHALF);
  float ang = (float)pos * inv;
  float c = __cosf(ang), s = __sinf(ang);
  float* base = x + (size_t)tt * DMODEL + hh * HEADDIM;
  float x1 = base[i], x2 = base[i + RHALF];
  base[i]         = x1 * c - x2 * s;
  base[i + RHALF] = x1 * s + x2 * c;
}

// ---------------------------------------------------------------------------
// Causal attention, online softmax, one wave32 per (b, head, qpos)
// ---------------------------------------------------------------------------
__global__ void attn_kernel(const float* __restrict__ q, const float* __restrict__ k,
                            const float* __restrict__ v, float* __restrict__ ctx) {
  int wid = (int)(((size_t)blockIdx.x * blockDim.x + threadIdx.x) >> 5);
  int lane = threadIdx.x & 31;
  if (wid >= BATCH * NHEAD * SEQ) return;
  int qpos = wid & (SEQ - 1);
  int hh = (wid >> 10) & (NHEAD - 1);
  int b = wid >> 14;
  size_t rowq = ((size_t)(b * SEQ + qpos)) * DMODEL + hh * HEADDIM;
  float q0 = q[rowq + lane] * 0.125f;          // 1/sqrt(64)
  float q1 = q[rowq + lane + 32] * 0.125f;
  float m = -3.0e38f, sum = 0.f, a0 = 0.f, a1 = 0.f;
  for (int j = 0; j <= qpos; ++j) {
    size_t rk = ((size_t)(b * SEQ + j)) * DMODEL + hh * HEADDIM;
    float d = q0 * k[rk + lane] + q1 * k[rk + lane + 32];
    d = waveSum(d);
    float mn = fmaxf(m, d);
    float sc = __expf(m - mn);
    float p = __expf(d - mn);
    sum = sum * sc + p;
    a0 = a0 * sc + p * v[rk + lane];
    a1 = a1 * sc + p * v[rk + lane + 32];
    m = mn;
  }
  float inv = 1.0f / sum;
  ctx[rowq + lane] = a0 * inv;
  ctx[rowq + lane + 32] = a1 * inv;
}

// ---------------------------------------------------------------------------
// Router: top-2 softmax gates (dense [T,E]); one wave32 per token
// ---------------------------------------------------------------------------
__global__ void router_kernel(const unsigned short* __restrict__ xn,
                              const float* __restrict__ rw,
                              float* __restrict__ gates) {
  int wave = threadIdx.x >> 5, lane = threadIdx.x & 31;
  int tt = blockIdx.x * (blockDim.x >> 5) + wave;
  if (tt >= NTOK) return;
  const unsigned short* xr = xn + (size_t)tt * DMODEL;
  float lg[NEXP];
#pragma unroll
  for (int e = 0; e < NEXP; ++e) {
    const float* we = rw + (size_t)e * DMODEL;
    float acc = 0.f;
    for (int d = lane; d < DMODEL; d += 32) acc += bf2f(xr[d]) * we[d];
    lg[e] = waveSum(acc);
  }
  if (lane == 0) {
    int bi = 0; float bv = lg[0];
    for (int e = 1; e < NEXP; ++e) if (lg[e] > bv) { bv = lg[e]; bi = e; }
    int si = 0; float sv = -3.0e38f;
    for (int e = 0; e < NEXP; ++e)
      if (e != bi && lg[e] > sv) { sv = lg[e]; si = e; }
    float w2g = __expf(sv - bv);
    float inv = 1.0f / (1.0f + w2g);
    float* g = gates + (size_t)tt * NEXP;
    for (int e = 0; e < NEXP; ++e) g[e] = 0.f;
    g[bi] = inv;
    g[si] = w2g * inv;
  }
}

// ---------------------------------------------------------------------------
// h_bf16 = bf16( silu(h1) * h3 )
// ---------------------------------------------------------------------------
__global__ void silu_mul_kernel(const float* __restrict__ h1,
                                const float* __restrict__ h3,
                                unsigned short* __restrict__ h, size_t n) {
  size_t i = (size_t)blockIdx.x * blockDim.x + threadIdx.x;
  size_t stride = (size_t)gridDim.x * blockDim.x;
  for (; i < n; i += stride) {
    float a = h1[i];
    float s = a / (1.0f + __expf(-a));
    h[i] = f2bf(s * h3[i]);
  }
}

// ---------------------------------------------------------------------------
// Host orchestration
// ---------------------------------------------------------------------------
extern "C" void kernel_launch(void* const* d_in, const int* in_sizes, int n_in,
                              void* d_out, int out_size, void* d_ws, size_t ws_size,
                              hipStream_t stream) {
  (void)in_sizes; (void)n_in; (void)out_size; (void)ws_size;
  const int*   tokens       = (const int*)d_in[0];
  const float* embed        = (const float*)d_in[1];
  const float* attn_norm_w  = (const float*)d_in[2];
  const float* wq           = (const float*)d_in[3];
  const float* wk           = (const float*)d_in[4];
  const float* wv           = (const float*)d_in[5];
  const float* wo           = (const float*)d_in[6];
  const float* moe_norm_w   = (const float*)d_in[7];
  const float* router_w     = (const float*)d_in[8];
  const float* w1           = (const float*)d_in[9];
  const float* w2           = (const float*)d_in[10];
  const float* w3           = (const float*)d_in[11];
  const float* final_norm_w = (const float*)d_in[12];
  float* out = (float*)d_out;

  char* ws = (char*)d_ws;
  size_t off = 0;
  auto alloc = [&](size_t bytes) -> void* {
    void* p = ws + off;
    off += (bytes + 255) & ~(size_t)255;
    return p;
  };
  const size_t nAttnW = (size_t)NLAYER * DMODEL * DMODEL;
  const size_t nFfnW  = (size_t)NLAYER * NEXP * DMODEL * HIDDEN;

  unsigned short* eb_bf  = (unsigned short*)alloc((size_t)VOCAB * DMODEL * 2);
  unsigned short* wq_bf  = (unsigned short*)alloc(nAttnW * 2);
  unsigned short* wk_bf  = (unsigned short*)alloc(nAttnW * 2);
  unsigned short* wv_bf  = (unsigned short*)alloc(nAttnW * 2);
  unsigned short* wo_bf  = (unsigned short*)alloc(nAttnW * 2);
  unsigned short* w1_bf  = (unsigned short*)alloc(nFfnW * 2);
  unsigned short* w2_bf  = (unsigned short*)alloc(nFfnW * 2);
  unsigned short* w3_bf  = (unsigned short*)alloc(nFfnW * 2);
  float*          x      = (float*)alloc((size_t)NTOK * DMODEL * 4);
  unsigned short* xn_bf  = (unsigned short*)alloc((size_t)NTOK * DMODEL * 2);
  float*          qf     = (float*)alloc((size_t)NTOK * DMODEL * 4);
  float*          kf     = (float*)alloc((size_t)NTOK * DMODEL * 4);
  float*          vf     = (float*)alloc((size_t)NTOK * DMODEL * 4);
  float*          ctxf   = (float*)alloc((size_t)NTOK * DMODEL * 4);
  unsigned short* ctx_bf = (unsigned short*)alloc((size_t)NTOK * DMODEL * 2);
  float*          h1f    = (float*)alloc((size_t)NTOK * HIDDEN * 4);
  float*          h3f    = (float*)alloc((size_t)NTOK * HIDDEN * 4);
  unsigned short* h_bf   = (unsigned short*)alloc((size_t)NTOK * HIDDEN * 2);
  float*          gates  = (float*)alloc((size_t)NTOK * NEXP * 4);

  auto cast = [&](const float* s, unsigned short* d, size_t n) {
    int blocks = (int)((n + 255) / 256);
    if (blocks > 262144) blocks = 262144;
    cast_bf16_kernel<<<blocks, 256, 0, stream>>>(s, d, n);
  };
  auto gemm = [&](int mode, bool transb, const unsigned short* A,
                  const unsigned short* Bm, float* C, int M, int N, int K,
                  const float* scale, int sstride) {
    dim3 grid(N / 128, M / 64);
    if (transb)
      gemm_wmma_kernel<0, true><<<grid, 256, 0, stream>>>(A, Bm, C, M, N, K, scale, sstride);
    else if (mode == 0)
      gemm_wmma_kernel<0, false><<<grid, 256, 0, stream>>>(A, Bm, C, M, N, K, scale, sstride);
    else if (mode == 1)
      gemm_wmma_kernel<1, false><<<grid, 256, 0, stream>>>(A, Bm, C, M, N, K, scale, sstride);
    else
      gemm_wmma_kernel<2, false><<<grid, 256, 0, stream>>>(A, Bm, C, M, N, K, scale, sstride);
  };

  // --- weight casts (redone every call: deterministic) ---
  cast(embed, eb_bf, (size_t)VOCAB * DMODEL);
  cast(wq, wq_bf, nAttnW);
  cast(wk, wk_bf, nAttnW);
  cast(wv, wv_bf, nAttnW);
  cast(wo, wo_bf, nAttnW);
  cast(w1, w1_bf, nFfnW);
  cast(w2, w2_bf, nFfnW);
  cast(w3, w3_bf, nFfnW);

  // --- x = embed[tokens] ---
  embed_gather_kernel<<<NTOK, 256, 0, stream>>>(tokens, embed, x);

  const int ropeThreads = NTOK * NHEAD * RHALF;     // 1,048,576
  const int attnThreads = BATCH * NHEAD * SEQ * 32; // 1,048,576

  for (int l = 0; l < NLAYER; ++l) {
    const size_t aw = (size_t)l * DMODEL * DMODEL;
    // attention block
    rmsnorm_bf16_kernel<<<NTOK, 256, 0, stream>>>(x, attn_norm_w + (size_t)l * DMODEL, xn_bf);
    gemm(0, false, xn_bf, wq_bf + aw, qf, NTOK, DMODEL, DMODEL, nullptr, 0);
    gemm(0, false, xn_bf, wk_bf + aw, kf, NTOK, DMODEL, DMODEL, nullptr, 0);
    gemm(0, false, xn_bf, wv_bf + aw, vf, NTOK, DMODEL, DMODEL, nullptr, 0);
    rope_kernel<<<ropeThreads / 256, 256, 0, stream>>>(qf);
    rope_kernel<<<ropeThreads / 256, 256, 0, stream>>>(kf);
    attn_kernel<<<attnThreads / 256, 256, 0, stream>>>(qf, kf, vf, ctxf);
    cast(ctxf, ctx_bf, (size_t)NTOK * DMODEL);
    gemm(1, false, ctx_bf, wo_bf + aw, x, NTOK, DMODEL, DMODEL, nullptr, 0); // x += ctx@wo

    // MoE block (dense over experts, matching the reference loop)
    rmsnorm_bf16_kernel<<<NTOK, 256, 0, stream>>>(x, moe_norm_w + (size_t)l * DMODEL, xn_bf);
    router_kernel<<<NTOK / 8, 256, 0, stream>>>(xn_bf, router_w + (size_t)l * NEXP * DMODEL, gates);
    for (int e = 0; e < NEXP; ++e) {
      const size_t fw = ((size_t)l * NEXP + e) * (size_t)DMODEL * HIDDEN;
      gemm(0, false, xn_bf, w1_bf + fw, h1f, NTOK, HIDDEN, DMODEL, nullptr, 0);
      gemm(0, false, xn_bf, w3_bf + fw, h3f, NTOK, HIDDEN, DMODEL, nullptr, 0);
      silu_mul_kernel<<<16384, 256, 0, stream>>>(h1f, h3f, h_bf, (size_t)NTOK * HIDDEN);
      // x += (h @ w2[e]) * gates[:, e]
      gemm(2, false, h_bf, w2_bf + fw, x, NTOK, DMODEL, HIDDEN, gates + e, NEXP);
    }
  }

  // --- final norm + logits = xn @ embed.T ---
  rmsnorm_bf16_kernel<<<NTOK, 256, 0, stream>>>(x, final_norm_w, xn_bf);
  gemm(0, true, xn_bf, eb_bf, out, NTOK, VOCAB, DMODEL, nullptr, 0);
}
